// CATSCluster_39891656245254
// MI455X (gfx1250) — matrix operand order repaired and checked by reading, not compiled, not run on hardware
//
#include <hip/hip_runtime.h>
#include <hip/hip_bf16.h>
#include <math.h>

#define EMB 768

typedef __attribute__((ext_vector_type(16))) _Float16 v16h;
typedef __attribute__((ext_vector_type(8)))  _Float16 v8h;
typedef __attribute__((ext_vector_type(8)))  float    v8f;

#define WAVES      4       // wave32 waves per block
#define BLOCK_T    128     // threads per block
#define BLOCK_ROWS 64      // rows per block (16 per wave)
#define SB_STRIDE  40      // halves per staged-B row (32 + 8 pad; 80B rows, 16B aligned)
#define SB_BUF_H   (BLOCK_T * SB_STRIDE)      // halves per staging buffer (5120)
#define SB_BUF_B   (SB_BUF_H * 2)             // bytes per staging buffer (10240)
#define SH_STRIDE  264     // halves per hidden row (256 + 8 pad)

// f16 weight workspace layout (element offsets into d_ws as _Float16*)
#define W1_ELEMS (256 * 768)
#define W2_ELEMS (128 * 256)
#define OFF_W1 0
#define OFF_W2 (W1_ELEMS)                       // 196608
#define OFF_W3 (W1_ELEMS + W2_ELEMS)            // 229376
#define OFF_W4 (2 * W1_ELEMS + W2_ELEMS)        // 425984
#define W_TOTAL (2 * W1_ELEMS + 2 * W2_ELEMS)   // 458752 halves = 917504 B

// ---------------- CDNA5 async memory->LDS copy (ASYNCcnt tracked) ----------------
__device__ __forceinline__ void async_cp16(unsigned lds_off, const _Float16* g) {
  // LDS[lds_off .. +15] = MEM[g .. +15], no VGPR data, tracked by ASYNCcnt
  asm volatile("global_load_async_to_lds_b128 %0, %1, off"
               :: "v"(lds_off), "v"(g) : "memory");
}
__device__ __forceinline__ void wait_async0() {
  asm volatile("s_wait_asynccnt 0" ::: "memory");
}
__device__ __forceinline__ void wait_async4() {
  asm volatile("s_wait_asynccnt 4" ::: "memory");   // newest 4 (next chunk) may remain
}

// Issue the async stage of a 32(K) x 128(N) f16 weight chunk into an LDS buffer.
// W16 is row-major [out][in] f16; thread t owns output column c0+t (4 x 16B pieces).
__device__ __forceinline__ void issue_stage(unsigned sb_off, const _Float16* W16,
                                            int ldw, int c0, int k0) {
  const int t = threadIdx.x;                         // 0..127
  const _Float16* src = W16 + (size_t)(c0 + t) * ldw + k0;
  const unsigned dst = sb_off + (unsigned)t * (SB_STRIDE * 2);
#pragma unroll
  for (int i = 0; i < 4; ++i)
    async_cp16(dst + i * 16, src + i * 8);
}

// ---- A fragment from global f32 rows (16-bit A 16x32 layout, ISA 7.12.2) ----
// lanes 0-15 : K = {0..7, 16..23}; lanes 16-31 : K = {8..15, 24..31}; kbase = k0 + hs*8
__device__ __forceinline__ v16h load_a_global(const float* rowf, int kbase) {
  const float4 l0 = *(const float4*)(rowf + kbase);
  const float4 l1 = *(const float4*)(rowf + kbase + 4);
  const float4 h0 = *(const float4*)(rowf + kbase + 16);
  const float4 h1 = *(const float4*)(rowf + kbase + 20);
  v16h a;
  a[0]=(_Float16)l0.x;  a[1]=(_Float16)l0.y;  a[2]=(_Float16)l0.z;  a[3]=(_Float16)l0.w;
  a[4]=(_Float16)l1.x;  a[5]=(_Float16)l1.y;  a[6]=(_Float16)l1.z;  a[7]=(_Float16)l1.w;
  a[8]=(_Float16)h0.x;  a[9]=(_Float16)h0.y;  a[10]=(_Float16)h0.z; a[11]=(_Float16)h0.w;
  a[12]=(_Float16)h1.x; a[13]=(_Float16)h1.y; a[14]=(_Float16)h1.z; a[15]=(_Float16)h1.w;
  return a;
}

// B fragment: 8 halves at +0, 8 halves at +16 (same K order as A)
__device__ __forceinline__ v16h load_h16(const _Float16* p) {
  union { v16h v; v8h h[2]; } u;
  u.h[0] = *(const v8h*)(p);
  u.h[1] = *(const v8h*)(p + 16);
  return u.v;
}

// layer-2 A fragment from LDS hidden slice: 8 halves at +0, 8 halves at +16
__device__ __forceinline__ v16h load_a_lds(const _Float16* p) {
  union { v16h v; v8h h[2]; } u;
  u.h[0] = *(const v8h*)(p);
  u.h[1] = *(const v8h*)(p + 16);
  return u.v;
}

// One 2-layer MLP for this wave's 16-row tile:
//   relu(relu(x[16x768] @ Wa^T) @ Wb^T) -> zout (16x128, C-fragment layout)
// Weight chunks are double-buffered via async-to-LDS DMA.
__device__ __forceinline__ void mlp2_tile(const float* rowf,
                                          const _Float16* Wa16, const _Float16* Wb16,
                                          _Float16* sB, unsigned sb_off,
                                          _Float16* sHw, int hs, int lrow,
                                          v8f zout[8]) {
  // ---------- layer 1: 16x768 @ 768x256 -> relu -> f16 hidden in LDS ----------
  for (int nh = 0; nh < 2; ++nh) {                   // two 128-col halves of 256 outputs
    v8f acc[8] = {};
    issue_stage(sb_off, Wa16, EMB, nh * 128, 0);     // prologue: chunk 0 -> buf0
    for (int kc = 0; kc < 24; ++kc) {
      const int cur = kc & 1;
      if (kc + 1 < 24) {
        issue_stage(sb_off + (cur ^ 1) * SB_BUF_B, Wa16, EMB, nh * 128, (kc + 1) * 32);
        wait_async4();                               // chunk kc landed (mine)
      } else {
        wait_async0();
      }
      __syncthreads();                               // chunk kc landed (all waves)
      v16h a = load_a_global(rowf, kc * 32 + hs * 8);
      const _Float16* bb = sB + cur * SB_BUF_H;
#pragma unroll
      for (int nt = 0; nt < 8; ++nt) {
        v16h b = load_h16(bb + (nt * 16 + lrow) * SB_STRIDE + hs * 8);
        acc[nt] = __builtin_amdgcn_wmma_f32_16x16x32_f16(
            false, a, false, b, (short)0, acc[nt], false, false);
      }
      __syncthreads();                               // all reads of `cur` done before reuse
    }
    // ReLU + convert, write hidden cols [nh*128, nh*128+128) for this wave's rows
#pragma unroll
    for (int nt = 0; nt < 8; ++nt)
#pragma unroll
      for (int v = 0; v < 8; ++v) {
        float x = acc[nt][v];
        x = x > 0.f ? x : 0.f;
        sHw[(v + 8 * hs) * SH_STRIDE + nh * 128 + nt * 16 + lrow] = (_Float16)x;
      }
  }

  // ---------- layer 2: 16x256 @ 256x128 -> relu -> regs ----------
  v8f acc2[8] = {};
  issue_stage(sb_off, Wb16, 256, 0, 0);              // prologue: chunk 0 -> buf0
  for (int kc = 0; kc < 8; ++kc) {
    const int cur = kc & 1;
    if (kc + 1 < 8) {
      issue_stage(sb_off + (cur ^ 1) * SB_BUF_B, Wb16, 256, 0, (kc + 1) * 32);
      wait_async4();
    } else {
      wait_async0();
    }
    __syncthreads();
    v16h a = load_a_lds(sHw + lrow * SH_STRIDE + kc * 32 + hs * 8);
    const _Float16* bb = sB + cur * SB_BUF_H;
#pragma unroll
    for (int nt = 0; nt < 8; ++nt) {
      v16h b = load_h16(bb + (nt * 16 + lrow) * SB_STRIDE + hs * 8);
      acc2[nt] = __builtin_amdgcn_wmma_f32_16x16x32_f16(
          false, a, false, b, (short)0, acc2[nt], false, false);
    }
    __syncthreads();
  }
#pragma unroll
  for (int nt = 0; nt < 8; ++nt)
#pragma unroll
    for (int v = 0; v < 8; ++v) {
      float x = acc2[nt][v];
      zout[nt][v] = x > 0.f ? x : 0.f;
    }
}

// ---------------- pre-pass: convert W1..W4 to f16 in workspace ----------------
__global__ void __launch_bounds__(256)
convert_weights(const float* __restrict__ W1, const float* __restrict__ W2,
                const float* __restrict__ W3, const float* __restrict__ W4,
                _Float16* __restrict__ ws) {
  int i = blockIdx.x * blockDim.x + threadIdx.x;
  const int stride = gridDim.x * blockDim.x;
  for (; i < W_TOTAL; i += stride) {
    float v;
    if (i < OFF_W2)      v = W1[i - OFF_W1];
    else if (i < OFF_W3) v = W2[i - OFF_W2];
    else if (i < OFF_W4) v = W3[i - OFF_W3];
    else                 v = W4[i - OFF_W4];
    ws[i] = (_Float16)v;
  }
}

// ---------------- main fused kernel ----------------
__global__ void __launch_bounds__(BLOCK_T)
cats_mlp_kernel(const float* __restrict__ X, const _Float16* __restrict__ Wh,
                const float* __restrict__ W5, float* __restrict__ out) {
  __shared__ _Float16 sB[2 * SB_BUF_H];                 // 20 KB double-buffered weights
  __shared__ _Float16 sH[WAVES * 16 * SH_STRIDE];       // 33 KB hidden / reused for Xpqz

  const int tid  = threadIdx.x;
  const int lane = tid & 31;
  const int wave = tid >> 5;
  const int lrow = lane & 15;   // row (A) or col (B/C) within 16
  const int hs   = lane >> 4;   // half-wave select

  // global row handled by this lane's M-tile position
  const int r = blockIdx.x * BLOCK_ROWS + wave * 16 + lrow;
  const int n = r >> 12;                 // r / 4096
  const int m = r & 4095;                // r % 4096
  const float* rowf = X + ((size_t)n * 4097 + 1 + m) * (3 * EMB);

  _Float16* sHw = sH + wave * 16 * SH_STRIDE;
  const unsigned sb_off = (unsigned)(size_t)sB;         // LDS byte offset of sB

  const _Float16* W1h = Wh + OFF_W1;
  const _Float16* W2h = Wh + OFF_W2;
  const _Float16* W3h = Wh + OFF_W3;
  const _Float16* W4h = Wh + OFF_W4;

  v8f z[8], t[8];
  mlp2_tile(rowf + EMB,     W1h, W2h, sB, sb_off, sHw, hs, lrow, z);   // Xp1z
  mlp2_tile(rowf + 2 * EMB, W1h, W2h, sB, sb_off, sHw, hs, lrow, t);   // Xp2z
#pragma unroll
  for (int nt = 0; nt < 8; ++nt)
#pragma unroll
    for (int v = 0; v < 8; ++v)
      z[nt][v] = fabsf(z[nt][v] - t[nt][v]);                           // |Xp1z - Xp2z|
  mlp2_tile(rowf,           W3h, W4h, sB, sb_off, sHw, hs, lrow, t);   // Xqz

  // All waves done reading hidden slices before the buffer is repurposed.
  __syncthreads();
  float* sX = (float*)sH;                                              // 64 x 128 f32
#pragma unroll
  for (int nt = 0; nt < 8; ++nt)
#pragma unroll
    for (int v = 0; v < 8; ++v) {
      int row = wave * 16 + v + 8 * hs;
      sX[row * 128 + nt * 16 + lrow] = t[nt][v] * z[nt][v];            // Xpqz
    }
  __syncthreads();

  // Final: tanh(relu(Xpqz @ W5^T)), one row per thread
  if (tid < BLOCK_ROWS) {
    const float* xr = sX + tid * 128;
    float s = 0.f;
#pragma unroll 8
    for (int c = 0; c < 128; ++c) s = fmaf(xr[c], W5[c], s);
    s = s > 0.f ? s : 0.f;
    out[blockIdx.x * BLOCK_ROWS + tid] = tanhf(s);
  }
}

extern "C" void kernel_launch(void* const* d_in, const int* in_sizes, int n_in,
                              void* d_out, int out_size, void* d_ws, size_t ws_size,
                              hipStream_t stream) {
  const float* X  = (const float*)d_in[0];
  const float* W1 = (const float*)d_in[1];
  const float* W2 = (const float*)d_in[2];
  const float* W3 = (const float*)d_in[3];
  const float* W4 = (const float*)d_in[4];
  const float* W5 = (const float*)d_in[5];
  float* out = (float*)d_out;
  _Float16* Wh = (_Float16*)d_ws;        // needs W_TOTAL*2 = 917,504 bytes

  // Pre-pass: f32 -> f16 weight conversion into workspace (tiny; 0.9 MB)
  hipLaunchKernelGGL(convert_weights, dim3(448), dim3(256), 0, stream,
                     W1, W2, W3, W4, Wh);

  const int totalRows = 16 * 4096;                 // 65,536
  dim3 grid(totalRows / BLOCK_ROWS);               // 1024 blocks
  dim3 block(BLOCK_T);                             // 128 threads = 4 wave32
  hipLaunchKernelGGL(cats_mlp_kernel, grid, block, 0, stream,
                     X, Wh, W5, out);
}
